// SABlock_69398081569489
// MI455X (gfx1250) — matrix operand order repaired and checked by reading, not compiled, not run on hardware
//
#include <hip/hip_runtime.h>

typedef __bf16 bf16;
typedef __attribute__((ext_vector_type(4)))  __bf16 v4bf;
typedef __attribute__((ext_vector_type(8)))  __bf16 v8bf;
typedef __attribute__((ext_vector_type(16))) __bf16 v16bf;
typedef __attribute__((ext_vector_type(8)))  float  v8f;

// ---------------------------------------------------------------- constants
#define Cc   768
#define NH   12
#define HD   64
#define BW   128         // B * 4x4 windows
#define Tt   256         // tokens per window
#define Mrows 32768      // BW * Tt
#define QKV_N 2304       // 3*C
#define SCALE 0.125f     // 64^-0.5

// ---------------------------------------------------------------- CDNA5 async copy
// GLOBAL_LOAD_ASYNC_TO_LDS_B128, GVS mode: uniform SGPR base + per-lane offset.
// Tracked by ASYNCcnt -> must s_wait_asynccnt 0 before the barrier.
__device__ __forceinline__ void async_b128(void* lds_dst, const void* sbase, unsigned voff) {
  unsigned lds = (unsigned)(unsigned long long)lds_dst;   // low 32b of flat = LDS offset
  asm volatile("global_load_async_to_lds_b128 %0, %1, %2"
               :: "v"(lds), "v"(voff), "s"(sbase) : "memory");
}
__device__ __forceinline__ void wait_async0() {
  asm volatile("s_wait_asynccnt 0x0" ::: "memory");
}

// ---------------------------------------------------------------- WMMA utils
__device__ __forceinline__ v8f wmma_bf16(v16bf a, v16bf b, v8f c) {
  return __builtin_amdgcn_wmma_f32_16x16x32_bf16(false, a, false, b,
                                                 (short)0, c, false, false);
}

// A fragment: 16x32 bf16, row-major LDS tile (CDNA5 ISA 16-bit A layout).
__device__ __forceinline__ v16bf frag_a(const bf16* base, int stride, int lane, int kBase) {
  const bf16* p = base + (lane & 15) * stride + kBase + ((lane >> 4) << 3);
  v8bf lo = *(const v8bf*)p;
  v8bf hi = *(const v8bf*)(p + 16);
  return __builtin_shufflevector(lo, hi, 0,1,2,3,4,5,6,7,8,9,10,11,12,13,14,15);
}
// B fragment: 32x16 bf16 stored [n][k] in LDS.
__device__ __forceinline__ v16bf frag_b(const bf16* base, int stride, int lane, int kBase) {
  const bf16* p = base + (lane & 15) * stride + kBase + ((lane >> 4) << 3);
  v8bf lo = *(const v8bf*)p;
  v8bf hi = *(const v8bf*)(p + 16);
  return __builtin_shufflevector(lo, hi, 0,1,2,3,4,5,6,7,8,9,10,11,12,13,14,15);
}

// ---------------------------------------------------------------- stage 0: converts
__global__ void k_conv_x(const float* __restrict__ X, bf16* __restrict__ Xw) {
  size_t i = ((size_t)blockIdx.x * 256 + threadIdx.x) * 4;
  if (i >= (size_t)Mrows * Cc) return;
  int m = (int)(i / Cc), c = (int)(i % Cc);
  int bw = m >> 8, tt = m & 255;
  int b  = bw >> 4, wi = bw & 15;
  int wh = wi >> 2, ww = wi & 3;
  int r  = tt >> 4, cl = tt & 15;
  int hw = (wh * 16 + r) * 64 + ww * 16 + cl;
  const float4 v = *(const float4*)(X + ((size_t)b * 4096 + hw) * Cc + c);
  v4bf o = { (bf16)v.x, (bf16)v.y, (bf16)v.z, (bf16)v.w };
  *(v4bf*)(Xw + i) = o;
}

__global__ void k_conv_w(const float* __restrict__ src, bf16* __restrict__ dst, int n4) {
  int i = blockIdx.x * 256 + threadIdx.x;
  if (i >= n4) return;
  const float4 v = *(const float4*)(src + (size_t)i * 4);
  v4bf o = { (bf16)v.x, (bf16)v.y, (bf16)v.z, (bf16)v.w };
  *(v4bf*)(dst + (size_t)i * 4) = o;
}

// ---------------------------------------------------------------- GEMM core
// 128x128 block, K-step 32, 8 waves each 64x32, async double-buffered LDS.
#define LSTR 40
#define TILE (128 * LSTR)

// stage one 128x32 bf16 tile (rows stride Cc in global) into LDS via async copies
__device__ __forceinline__ void tile_async(bf16* dst, const bf16* src, int tid) {
  #pragma unroll
  for (int i = tid; i < 512; i += 256) {
    int r = i >> 2, c = (i & 3) * 8;
    async_b128(dst + r * LSTR + c, src, (unsigned)((r * Cc + c) * 2));
  }
}

// computes acc += A_tile(wave) x B_tile(wave) for one K-step
__device__ __forceinline__ void gemm_step(const bf16* As, const bf16* Bs,
                                          int wm, int wn, int lane, v8f acc[4][2]) {
  v16bf a[4], b[2];
  #pragma unroll
  for (int i = 0; i < 4; ++i) a[i] = frag_a(As + (wm + i * 16) * LSTR, LSTR, lane, 0);
  #pragma unroll
  for (int j = 0; j < 2; ++j) b[j] = frag_b(Bs + (wn + j * 16) * LSTR, LSTR, lane, 0);
  #pragma unroll
  for (int i = 0; i < 4; ++i)
    #pragma unroll
    for (int j = 0; j < 2; ++j)
      acc[i][j] = wmma_bf16(a[i], b[j], acc[i][j]);
}

// ---------------------------------------------------------------- stage 1: QKV GEMM
__global__ void k_qkv(const bf16* __restrict__ X, const bf16* __restrict__ Wq,
                      bf16* __restrict__ Q, bf16* __restrict__ K,
                      bf16* __restrict__ Vt) {
  __shared__ __align__(16) bf16 As[2 * TILE];
  __shared__ __align__(16) bf16 Bs[2 * TILE];
  const int mBase = blockIdx.x * 128;
  const int nBase = blockIdx.y * 128;
  const int tid = threadIdx.x, lane = tid & 31, w = tid >> 5;
  const int wm = (w & 1) * 64, wn = (w >> 1) * 32;
  const bf16* aSrc = X  + (size_t)mBase * Cc;
  const bf16* bSrc = Wq + (size_t)nBase * Cc;
  v8f acc[4][2] = {};

  tile_async(As, aSrc, tid);
  tile_async(Bs, bSrc, tid);
  wait_async0();
  __syncthreads();

  int cur = 0;
  for (int k0 = 0; k0 < Cc; k0 += 32) {
    int nxt = cur ^ 1;
    if (k0 + 32 < Cc) {                       // prefetch next K-slice
      tile_async(As + nxt * TILE, aSrc + k0 + 32, tid);
      tile_async(Bs + nxt * TILE, bSrc + k0 + 32, tid);
    }
    gemm_step(As + cur * TILE, Bs + cur * TILE, wm, wn, lane, acc);
    wait_async0();
    __syncthreads();
    cur = nxt;
  }

  const int half = lane >> 4, nOff = lane & 15;
  #pragma unroll
  for (int i = 0; i < 4; ++i)
    #pragma unroll
    for (int j = 0; j < 2; ++j)
      #pragma unroll
      for (int r = 0; r < 8; ++r) {
        int m = mBase + wm + i * 16 + r + 8 * half;
        int n = nBase + wn + j * 16 + nOff;
        float val = acc[i][j][r];
        int bw = m >> 8, tt = m & 255;
        int seg = n / Cc, rem = n % Cc;
        int head = rem >> 6, d = rem & 63;
        size_t bh = (size_t)(bw * NH + head);
        if (seg == 0)      Q[(bh * Tt + tt) * HD + d] = (bf16)(val * SCALE);
        else if (seg == 1) K[(bh * Tt + tt) * HD + d] = (bf16)val;
        else               Vt[(bh * HD + d) * Tt + tt] = (bf16)val;
      }
}

// ---------------------------------------------------------------- stage 2: attention
#define QSTR 72
#define PSTR 264
__global__ void k_attn(const bf16* __restrict__ Q, const bf16* __restrict__ Kb,
                       const bf16* __restrict__ Vt,
                       const float* __restrict__ Rh, const float* __restrict__ Rw,
                       bf16* __restrict__ O) {
  extern __shared__ char smem[];
  bf16*  Sq    = (bf16*)smem;                 // 64 x QSTR
  bf16*  Sk    = Sq + 64 * QSTR;              // 256 x QSTR
  bf16*  Sv    = Sk + 256 * QSTR;             // 64 x PSTR  ([d][y])
  float* Ss    = (float*)(Sv + 64 * PSTR);    // 64 x 256
  bf16*  Sp    = (bf16*)(Ss + 64 * 256);      // 64 x PSTR
  float* Trh   = (float*)(Sp + 64 * PSTR);    // 64 x 16
  float* Trw   = Trh + 64 * 16;               // 64 x 16
  float* Spart = Trw + 64 * 16;               // 64 x 4
  float* Srow  = Spart + 256;                 // 64

  const int bh = blockIdx.x >> 2;
  const int qb = blockIdx.x & 3;
  const int head = bh % NH, bw = bh / NH;
  const int tid = threadIdx.x, lane = tid & 31, w = tid >> 5;

  const bf16* qg = Q  + ((size_t)bh * Tt + qb * 64) * HD;
  const bf16* kg = Kb + (size_t)bh * Tt * HD;
  const bf16* vg = Vt + (size_t)bh * HD * Tt;

  // ---- async stage Sq (64x64), Sk (256x64), Sv (64x256) into LDS
  #pragma unroll
  for (int i = tid; i < 512; i += 256) {
    int r = i >> 3, c = (i & 7) * 8;
    async_b128(Sq + r * QSTR + c, qg, (unsigned)((r * HD + c) * 2));
  }
  #pragma unroll
  for (int i = tid; i < 2048; i += 256) {
    int r = i >> 3, c = (i & 7) * 8;
    async_b128(Sk + r * QSTR + c, kg, (unsigned)((r * HD + c) * 2));
  }
  #pragma unroll
  for (int i = tid; i < 2048; i += 256) {
    int r = i >> 5, c = (i & 31) * 8;
    async_b128(Sv + r * PSTR + c, vg, (unsigned)((r * Tt + c) * 2));
  }
  wait_async0();
  __syncthreads();

  // ---- scores S = q @ k^T  (64x256): 64 tiles of 16x16, 8 per wave
  for (int tIdx = w * 8; tIdx < w * 8 + 8; ++tIdx) {
    int mt = tIdx >> 4, nt = tIdx & 15;
    v8f acc = {};
    #pragma unroll
    for (int k0 = 0; k0 < HD; k0 += 32) {
      v16bf a = frag_a(Sq + mt * 16 * QSTR, QSTR, lane, k0);
      v16bf b = frag_b(Sk + nt * 16 * QSTR, QSTR, lane, k0);
      acc = wmma_bf16(a, b, acc);
    }
    int mB = mt * 16 + 8 * (lane >> 4);
    int nB = nt * 16 + (lane & 15);
    #pragma unroll
    for (int r = 0; r < 8; ++r) Ss[(mB + r) * 256 + nB] = acc[r];
  }

  // ---- decomposed rel-pos tables (q pre-scaled by 0.125 -> undo with *8)
  for (int i = tid; i < 2048; i += 256) {
    int which = i >> 10;
    int x = (i >> 4) & 63, kk = i & 15;
    int gx = qb * 64 + x;
    int idx = which ? ((gx & 15) - kk + 15) : ((gx >> 4) - kk + 15);
    const float* rp = (which ? Rw : Rh) + idx * HD;
    const bf16* qr = Sq + x * QSTR;
    float s = 0.f;
    #pragma unroll 8
    for (int d = 0; d < HD; ++d) s += (float)qr[d] * rp[d];
    (which ? Trw : Trh)[x * 16 + kk] = s * 8.0f;
  }
  __syncthreads();

  // ---- softmax: 4 threads / row
  {
    int row = tid >> 2, part = tid & 3;
    float* sr = Ss + row * 256 + part * 64;
    float mx = -1e30f;
    #pragma unroll 8
    for (int j = 0; j < 64; ++j) {
      int y = part * 64 + j;
      float v = sr[j] + Trh[row * 16 + (y >> 4)] + Trw[row * 16 + (y & 15)];
      sr[j] = v;
      mx = fmaxf(mx, v);
    }
    Spart[row * 4 + part] = mx;
  }
  __syncthreads();
  if (tid < 64)
    Srow[tid] = fmaxf(fmaxf(Spart[tid * 4], Spart[tid * 4 + 1]),
                      fmaxf(Spart[tid * 4 + 2], Spart[tid * 4 + 3]));
  __syncthreads();
  {
    int row = tid >> 2, part = tid & 3;
    float mx = Srow[row];
    const float* sr = Ss + row * 256 + part * 64;
    bf16* pr = Sp + row * PSTR + part * 64;
    float sum = 0.f;
    #pragma unroll 8
    for (int j = 0; j < 64; ++j) {
      float e = __expf(sr[j] - mx);
      pr[j] = (bf16)e;
      sum += e;
    }
    Spart[row * 4 + part] = sum;
  }
  __syncthreads();
  if (tid < 64)
    Srow[tid] = 1.0f / (Spart[tid * 4] + Spart[tid * 4 + 1] +
                        Spart[tid * 4 + 2] + Spart[tid * 4 + 3]);
  __syncthreads();

  // ---- O = P @ V : 4x4 tiles, 2 per wave, K = 256
  {
    int mt = w >> 1;
    int ntb = (w & 1) * 2;
    v8f acc2[2] = {};
    for (int k0 = 0; k0 < 256; k0 += 32) {
      v16bf a = frag_a(Sp + mt * 16 * PSTR, PSTR, lane, k0);
      #pragma unroll
      for (int j = 0; j < 2; ++j) {
        v16bf b = frag_b(Sv + (ntb + j) * 16 * PSTR, PSTR, lane, k0);
        acc2[j] = wmma_bf16(a, b, acc2[j]);
      }
    }
    int mB = mt * 16 + 8 * (lane >> 4);
    #pragma unroll
    for (int j = 0; j < 2; ++j)
      #pragma unroll
      for (int r = 0; r < 8; ++r) {
        int m = mB + r;
        float val = acc2[j][r] * Srow[m];
        size_t row = (size_t)bw * Tt + qb * 64 + m;
        O[row * Cc + head * HD + (ntb + j) * 16 + (lane & 15)] = (bf16)val;
      }
  }
}

#define ATTN_LDS (64*QSTR*2 + 256*QSTR*2 + 64*PSTR*2 + 64*256*4 + 64*PSTR*2 + 64*16*4*2 + 256*4 + 64*4)

// ---------------------------------------------------------------- stage 3: out-proj
__global__ void k_oproj(const bf16* __restrict__ Oin, const bf16* __restrict__ Wo,
                        const float* __restrict__ bias, float* __restrict__ out) {
  __shared__ __align__(16) bf16 As[2 * TILE];
  __shared__ __align__(16) bf16 Bs[2 * TILE];
  const int mBase = blockIdx.x * 128;
  const int nBase = blockIdx.y * 128;
  const int tid = threadIdx.x, lane = tid & 31, w = tid >> 5;
  const int wm = (w & 1) * 64, wn = (w >> 1) * 32;
  const bf16* aSrc = Oin + (size_t)mBase * Cc;
  const bf16* bSrc = Wo  + (size_t)nBase * Cc;
  v8f acc[4][2] = {};

  tile_async(As, aSrc, tid);
  tile_async(Bs, bSrc, tid);
  wait_async0();
  __syncthreads();

  int cur = 0;
  for (int k0 = 0; k0 < Cc; k0 += 32) {
    int nxt = cur ^ 1;
    if (k0 + 32 < Cc) {
      tile_async(As + nxt * TILE, aSrc + k0 + 32, tid);
      tile_async(Bs + nxt * TILE, bSrc + k0 + 32, tid);
    }
    gemm_step(As + cur * TILE, Bs + cur * TILE, wm, wn, lane, acc);
    wait_async0();
    __syncthreads();
    cur = nxt;
  }

  const int half = lane >> 4, nOff = lane & 15;
  #pragma unroll
  for (int i = 0; i < 4; ++i)
    #pragma unroll
    for (int j = 0; j < 2; ++j)
      #pragma unroll
      for (int r = 0; r < 8; ++r) {
        int m = mBase + wm + i * 16 + r + 8 * half;
        int n = nBase + wn + j * 16 + nOff;
        float val = acc[i][j][r] + bias[n];
        int bw = m >> 8, tt = m & 255;
        int b_ = bw >> 4, wi = bw & 15;
        int wh = wi >> 2, ww = wi & 3;
        int rr = tt >> 4, cl = tt & 15;
        int hw = (wh * 16 + rr) * 64 + ww * 16 + cl;
        out[((size_t)b_ * 4096 + hw) * Cc + n] = val;
      }
}

// ---------------------------------------------------------------- launcher
extern "C" void kernel_launch(void* const* d_in, const int* in_sizes, int n_in,
                              void* d_out, int out_size, void* d_ws, size_t ws_size,
                              hipStream_t stream) {
  const float* x      = (const float*)d_in[0];
  const float* w_qkv  = (const float*)d_in[1];
  const float* w_out  = (const float*)d_in[2];
  const float* b_out  = (const float*)d_in[3];
  const float* rel_h  = (const float*)d_in[4];
  const float* rel_w  = (const float*)d_in[5];
  float* out = (float*)d_out;

  const size_t MC = (size_t)Mrows * Cc;
  bf16* xw = (bf16*)d_ws;
  bf16* wq = xw + MC;
  bf16* wo = wq + (size_t)QKV_N * Cc;
  bf16* qb = wo + (size_t)Cc * Cc;
  bf16* kb = qb + MC;
  bf16* vt = kb + MC;
  bf16* ob = vt + MC;

  k_conv_x<<<(int)(MC / 4 / 256), 256, 0, stream>>>(x, xw);
  k_conv_w<<<(QKV_N * Cc / 4 + 255) / 256, 256, 0, stream>>>(w_qkv, wq, QKV_N * Cc / 4);
  k_conv_w<<<(Cc * Cc / 4 + 255) / 256, 256, 0, stream>>>(w_out, wo, Cc * Cc / 4);

  k_qkv<<<dim3(Mrows / 128, QKV_N / 128), 256, 0, stream>>>(xw, wq, qb, kb, vt);
  k_attn<<<BW * NH * 4, 256, ATTN_LDS, stream>>>(qb, kb, vt, rel_h, rel_w, ob);
  k_oproj<<<dim3(Mrows / 128, Cc / 128), 256, 0, stream>>>(ob, wo, b_out, out);
}
